// SelfMatchingAttention_28217935135074
// MI455X (gfx1250) — matrix operand order repaired and checked by reading, not compiled, not run on hardware
//
#include <hip/hip_runtime.h>
#include <hip/hip_bf16.h>

// ---------------------------------------------------------------------------
// Types for WMMA fragments (gfx1250, wave32)
// ---------------------------------------------------------------------------
typedef __attribute__((ext_vector_type(16))) _Float16 v16h;
typedef __attribute__((ext_vector_type(8)))  _Float16 v8h;
typedef __attribute__((ext_vector_type(8)))  float    v8f;

#define B_  4
#define L_  512
#define H_  64
#define D2_ 128
#define D4_ 256

// ---------------------------------------------------------------------------
// f32 -> f16 conversion (weights + activations staged once for WMMA)
// ---------------------------------------------------------------------------
__global__ void cvt_f32_f16(const float* __restrict__ in,
                            _Float16* __restrict__ out, int n) {
    int i = blockIdx.x * blockDim.x + threadIdx.x;
    if (i < n) out[i] = (_Float16)in[i];
}

// ---------------------------------------------------------------------------
// WMMA GEMM: out[M,N] = A16[M,K] @ W16[N,K]^T (+ bias[N])
// One wave per 16x64 output tile (4 independent 16x16 accumulators), K-loop in
// steps of 32.  Per K-step: 1 A fragment (2 x 16B) reused by 4 WMMAs against
// 4 B fragments (4 x 32B) -> vmem:wmma ratio 1.5:1 and register-level A reuse.
// Fragment layouts (this code's k order, same for A and B):
//  A: lane half h in {0,1}; row m = lane&15;
//     elements 0..15 = K (k0 + 16h .. +15), contiguous along an A row
//  B: column n = lane&15; K = k0 + 16h .. +15 contiguous along a W row
//  C: VGPR r, lane half -> m = r + 8h, n = lane&15
// ---------------------------------------------------------------------------
__global__ void gemm_wmma_f16(const _Float16* __restrict__ A,
                              const _Float16* __restrict__ W,
                              const float* __restrict__ bias,
                              float* __restrict__ out,
                              int M, int N, int K) {
    const int wavesPerBlock = blockDim.x >> 5;
    const int wave = blockIdx.x * wavesPerBlock + (threadIdx.x >> 5);
    const int tilesN = N >> 6;                 // 64-wide wave tiles
    const int tileM = (wave / tilesN) << 4;
    const int tileN = (wave % tilesN) << 6;
    if (tileM >= M) return;

    const int lane = threadIdx.x & 31;
    const int half = lane >> 4;     // 0 or 1
    const int l16  = lane & 15;

    v8f acc0 = {}, acc1 = {}, acc2 = {}, acc3 = {};

    const _Float16* arow0 = A + (size_t)(tileM + l16) * K + half * 16;
    const _Float16* b0 = W + (size_t)(tileN + l16) * K + half * 16;
    const _Float16* b1 = b0 + (size_t)16 * K;
    const _Float16* b2 = b0 + (size_t)32 * K;
    const _Float16* b3 = b0 + (size_t)48 * K;

    for (int k0 = 0; k0 < K; k0 += 32) {
        const _Float16* arow = arow0 + k0;
        if (k0 + 32 < K) {
            __builtin_prefetch(arow + 32, 0, 1);   // global_prefetch_b8
            __builtin_prefetch(b0 + k0 + 32, 0, 1);
        }
        v16h a;
        *(v8h*)&a       = *(const v8h*)(arow);        // K +0..7   (or +16..23)
        *((v8h*)&a + 1) = *(const v8h*)(arow + 8);    // K +8..15  (or +24..31)
        v16h vb0 = *(const v16h*)(b0 + k0);
        v16h vb1 = *(const v16h*)(b1 + k0);
        v16h vb2 = *(const v16h*)(b2 + k0);
        v16h vb3 = *(const v16h*)(b3 + k0);
        acc0 = __builtin_amdgcn_wmma_f32_16x16x32_f16(false, a, false, vb0,
                                                      (short)0, acc0, false, false);
        acc1 = __builtin_amdgcn_wmma_f32_16x16x32_f16(false, a, false, vb1,
                                                      (short)0, acc1, false, false);
        acc2 = __builtin_amdgcn_wmma_f32_16x16x32_f16(false, a, false, vb2,
                                                      (short)0, acc2, false, false);
        acc3 = __builtin_amdgcn_wmma_f32_16x16x32_f16(false, a, false, vb3,
                                                      (short)0, acc3, false, false);
    }

    // store 4 sub-tiles; lanes 0-15 cover rows 0-7, lanes 16-31 rows 8-15
    const v8f accs[4] = {acc0, acc1, acc2, acc3};
#pragma unroll
    for (int s = 0; s < 4; ++s) {
        const int n0 = tileN + s * 16;
        const float bval = bias ? bias[n0 + l16] : 0.0f;
        float* obase = out + (size_t)tileM * N + n0 + l16;
#pragma unroll
        for (int r = 0; r < 8; ++r) {
            int m = r + half * 8;
            obase[(size_t)m * N] = accs[s][r] + bval;
        }
    }
}

// ---------------------------------------------------------------------------
// Fused attention: per (b,t) block computes
//   scores_j = v . tanh(t[b,t] + p[b,j]) ; softmax over j ; c = sum_j a_j emb[b,j]
// then emits xc = concat(emb[b,t], c) in f32 and f16.
// Transcendental-bound (134M tanh total) -> pure VALU kernel, scores in LDS.
// ---------------------------------------------------------------------------
__global__ void attn_fused(const float* __restrict__ emb,
                           const float* __restrict__ p,
                           const float* __restrict__ t,
                           const float* __restrict__ v,
                           float* __restrict__ xc,
                           _Float16* __restrict__ xc16) {
    const int b  = blockIdx.x / L_;
    const int ti = blockIdx.x % L_;
    __shared__ float tvec[D2_];
    __shared__ float vsh[D2_];
    __shared__ float sc[L_];
    __shared__ float red[256];

    const int tid = threadIdx.x;            // 256 threads = 8 waves
    if (tid < D2_) {
        tvec[tid] = t[((size_t)b * L_ + ti) * D2_ + tid];
        vsh[tid]  = v[tid];
    }
    __syncthreads();

    const int wave = tid >> 5, lane = tid & 31;
    // --- scores ---
    for (int j = wave; j < L_; j += 8) {
        const float* prow = p + ((size_t)b * L_ + j) * D2_;
        float s = 0.f;
#pragma unroll
        for (int d = lane; d < D2_; d += 32)
            s += vsh[d] * tanhf(tvec[d] + prow[d]);
        for (int off = 16; off > 0; off >>= 1)
            s += __shfl_down(s, off);
        if (lane == 0) sc[j] = s;
    }
    __syncthreads();

    // --- softmax (max / exp / sum) ---
    float m = -3.4e38f;
    for (int j = tid; j < L_; j += 256) m = fmaxf(m, sc[j]);
    red[tid] = m; __syncthreads();
    for (int s2 = 128; s2 > 0; s2 >>= 1) {
        if (tid < s2) red[tid] = fmaxf(red[tid], red[tid + s2]);
        __syncthreads();
    }
    m = red[0]; __syncthreads();
    float sum = 0.f;
    for (int j = tid; j < L_; j += 256) {
        float e = __expf(sc[j] - m);
        sc[j] = e;
        sum += e;
    }
    red[tid] = sum; __syncthreads();
    for (int s2 = 128; s2 > 0; s2 >>= 1) {
        if (tid < s2) red[tid] += red[tid + s2];
        __syncthreads();
    }
    const float inv = 1.0f / red[0];
    __syncthreads();

    // --- context: thread owns (d = tid&127, j-half = tid>>7) ---
    const int d = tid & (D2_ - 1), jh = tid >> 7;
    float cpart = 0.f;
    for (int j = jh * 256; j < jh * 256 + 256; ++j)
        cpart += sc[j] * emb[((size_t)b * L_ + j) * D2_ + d];
    red[tid] = cpart; __syncthreads();

    if (tid < D2_) {
        const float c  = (red[tid] + red[tid + 128]) * inv;
        const float e0 = emb[((size_t)b * L_ + ti) * D2_ + tid];
        const size_t base = ((size_t)b * L_ + ti) * D4_;
        xc[base + tid]        = e0;
        xc[base + D2_ + tid]  = c;
        xc16[base + tid]       = (_Float16)e0;
        xc16[base + D2_ + tid] = (_Float16)c;
    }
}

// ---------------------------------------------------------------------------
// Gated fusion: x16 = f16( sigmoid(g) * xc )   (GRU layer 0 consumes f16 only)
// ---------------------------------------------------------------------------
__global__ void gate_apply(const float* __restrict__ g,
                           const float* __restrict__ xc,
                           _Float16* __restrict__ x16, int n) {
    int i = blockIdx.x * blockDim.x + threadIdx.x;
    if (i < n) {
        float s = 1.0f / (1.0f + __expf(-g[i]));
        x16[i] = (_Float16)(s * xc[i]);
    }
}

// ---------------------------------------------------------------------------
// GRU recurrent scan. Input projections gi = x@Wih^T + bih were hoisted to the
// WMMA GEMM; only the h-recurrence runs here. One block per direction
// (gridDim.x == 2), Whh (192x64 f32 = 48KB) + h + gh resident in LDS.
//   phase A: gh[b,u] = h[b]·Whh[u] + bhh[u]   (768 dots of length 64)
//   phase B: gate math + h update (256 (b,j) pairs, one per thread)
// ---------------------------------------------------------------------------
__global__ void gru_rec(const float* __restrict__ gi_f,
                        const float* __restrict__ gi_b,
                        const float* __restrict__ Whh_f,
                        const float* __restrict__ Whh_b,
                        const float* __restrict__ bhh_f,
                        const float* __restrict__ bhh_b,
                        float* __restrict__ y,
                        _Float16* __restrict__ y16) {
    const int dir = blockIdx.x;
    const float* gi  = dir ? gi_b  : gi_f;
    const float* Whh = dir ? Whh_b : Whh_f;
    const float* bhh = dir ? bhh_b : bhh_f;

    __shared__ float sW[192 * 64];     // 48 KB
    __shared__ float h[B_ * H_];       // 1 KB
    __shared__ float gh[B_ * 192];     // 3 KB

    const int tid = threadIdx.x;       // 256
    for (int i = tid; i < 192 * 64; i += 256) sW[i] = Whh[i];
    h[tid] = 0.0f;
    __syncthreads();

    for (int step = 0; step < L_; ++step) {
        const int tstep = dir ? (L_ - 1 - step) : step;

        // phase A: recurrent GEMV into gh
        for (int idx = tid; idx < B_ * 192; idx += 256) {
            const int bb = idx / 192, u = idx % 192;
            const float* wrow = &sW[u * 64];
            const float* hrow = &h[bb * 64];
            float s = bhh[u];
#pragma unroll
            for (int k = 0; k < 64; ++k) s += hrow[k] * wrow[k];
            gh[idx] = s;
        }
        __syncthreads();

        // phase B: gates + state update
        const int bb = tid >> 6, j = tid & 63;
        const float* girow = gi + ((size_t)bb * L_ + tstep) * 192;
        const float gir = girow[j], giz = girow[64 + j], gin = girow[128 + j];
        const float ghr = gh[bb * 192 + j];
        const float ghz = gh[bb * 192 + 64 + j];
        const float ghn = gh[bb * 192 + 128 + j];
        const float r = 1.0f / (1.0f + __expf(-(gir + ghr)));
        const float z = 1.0f / (1.0f + __expf(-(giz + ghz)));
        const float n = tanhf(gin + r * ghn);
        const float hn = (1.0f - z) * n + z * h[tid];
        h[tid] = hn;
        const size_t yoff = ((size_t)bb * L_ + tstep) * (2 * H_) + dir * H_ + j;
        y[yoff]   = hn;
        y16[yoff] = (_Float16)hn;
        __syncthreads();
    }
}

// ---------------------------------------------------------------------------
// Host orchestration
// ---------------------------------------------------------------------------
extern "C" void kernel_launch(void* const* d_in, const int* in_sizes, int n_in,
                              void* d_out, int out_size, void* d_ws, size_t ws_size,
                              hipStream_t stream) {
    (void)in_sizes; (void)n_in; (void)out_size; (void)ws_size;

    const float* emb = (const float*)d_in[0];
    // d_in[1] = lengths (unused by forward)
    const float* Wp = (const float*)d_in[2];
    const float* Wt = (const float*)d_in[3];
    const float* Wg = (const float*)d_in[4];
    const float* v  = (const float*)d_in[5];
    // gru_params flattened: (layer,dir) major, then (Wih, Whh, bih, bhh)
    auto gp = [&](int ld, int which) -> const float* {
        return (const float*)d_in[6 + 4 * ld + which];
    };

    char* ws = (char*)d_ws;
    size_t off = 0;
    auto alloc = [&](size_t bytes) -> char* {
        char* pp = ws + off;
        off += (bytes + 255) & ~(size_t)255;
        return pp;
    };

    const int MT = B_ * L_;                 // 2048 token rows
    _Float16* emb16   = (_Float16*)alloc((size_t)MT * D2_ * 2);
    _Float16* Wp16    = (_Float16*)alloc((size_t)D2_ * D2_ * 2);
    _Float16* Wt16    = (_Float16*)alloc((size_t)D2_ * D2_ * 2);
    _Float16* Wg16    = (_Float16*)alloc((size_t)D4_ * D4_ * 2);
    _Float16* Wih16_0f = (_Float16*)alloc((size_t)192 * D4_ * 2);
    _Float16* Wih16_0b = (_Float16*)alloc((size_t)192 * D4_ * 2);
    _Float16* Wih16_1f = (_Float16*)alloc((size_t)192 * D2_ * 2);
    _Float16* Wih16_1b = (_Float16*)alloc((size_t)192 * D2_ * 2);
    float*    p_   = (float*)alloc((size_t)MT * D2_ * 4);
    float*    t_   = (float*)alloc((size_t)MT * D2_ * 4);
    float*    xc   = (float*)alloc((size_t)MT * D4_ * 4);
    _Float16* xc16 = (_Float16*)alloc((size_t)MT * D4_ * 2);
    float*    g    = (float*)alloc((size_t)MT * D4_ * 4);
    _Float16* x16  = (_Float16*)alloc((size_t)MT * D4_ * 2);
    float*    gi_f = (float*)alloc((size_t)MT * 192 * 4);
    float*    gi_b = (float*)alloc((size_t)MT * 192 * 4);
    float*    y    = (float*)alloc((size_t)MT * D2_ * 4);
    _Float16* y16  = (_Float16*)alloc((size_t)MT * D2_ * 2);
    _Float16* y16b = (_Float16*)alloc((size_t)MT * D2_ * 2);

    auto cvt = [&](const float* in, _Float16* o16, int n) {
        cvt_f32_f16<<<(n + 255) / 256, 256, 0, stream>>>(in, o16, n);
    };
    auto gemm = [&](const _Float16* A, const _Float16* W, const float* bias,
                    float* out, int M, int N, int K) {
        int waves = (M / 16) * (N / 64);       // one wave per 16x64 tile
        gemm_wmma_f16<<<(waves + 3) / 4, 128, 0, stream>>>(A, W, bias, out, M, N, K);
    };

    // Stage f16 copies (once; L2-resident afterwards)
    cvt(emb, emb16, MT * D2_);
    cvt(Wp, Wp16, D2_ * D2_);
    cvt(Wt, Wt16, D2_ * D2_);
    cvt(Wg, Wg16, D4_ * D4_);
    cvt(gp(0, 0), Wih16_0f, 192 * D4_);
    cvt(gp(1, 0), Wih16_0b, 192 * D4_);
    cvt(gp(2, 0), Wih16_1f, 192 * D2_);
    cvt(gp(3, 0), Wih16_1b, 192 * D2_);

    // Attention projections (WMMA)
    gemm(emb16, Wp16, nullptr, p_, MT, D2_, D2_);
    gemm(emb16, Wt16, nullptr, t_, MT, D2_, D2_);

    // Fused tanh-score / softmax / context / concat
    attn_fused<<<B_ * L_, 256, 0, stream>>>(emb, p_, t_, v, xc, xc16);

    // Gated fusion (WMMA + elementwise)
    gemm(xc16, Wg16, nullptr, g, MT, D4_, D4_);
    gate_apply<<<(MT * D4_ + 255) / 256, 256, 0, stream>>>(g, xc, x16, MT * D4_);

    // Layer 0: hoisted input projections (WMMA), then recurrent scan
    gemm(x16, Wih16_0f, gp(0, 2), gi_f, MT, 192, D4_);
    gemm(x16, Wih16_0b, gp(1, 2), gi_b, MT, 192, D4_);
    gru_rec<<<2, 256, 0, stream>>>(gi_f, gi_b, gp(0, 1), gp(1, 1),
                                   gp(0, 3), gp(1, 3), y, y16);

    // Layer 1
    gemm(y16, Wih16_1f, gp(2, 2), gi_f, MT, 192, D2_);
    gemm(y16, Wih16_1b, gp(3, 2), gi_b, MT, 192, D2_);
    gru_rec<<<2, 256, 0, stream>>>(gi_f, gi_b, gp(2, 1), gp(3, 1),
                                   gp(2, 3), gp(3, 3), (float*)d_out, y16b);
}
